// MultiHeadAttention_72937134621121
// MI455X (gfx1250) — matrix operand order repaired and checked by reading, not compiled, not run on hardware
//
#include <hip/hip_runtime.h>
#include <hip/hip_bf16.h>

#define D_MODEL 1024
#define N_HEADS 16
#define HEAD_DIM 64
#define BATCH 2
#define SEQ 2048
#define ROWS (BATCH * SEQ)   // 4096
#define LN_EPS 1e-5f
#define STILE 2052           // padded LDS row stride (floats) to break bank conflicts
#define MASK_LDS_OFF (16 * STILE * 4)          // byte offset of mask stripe in LDS
#define SCORES_LDS_BYTES (MASK_LDS_OFF + 16 * SEQ)  // 131328 + 32768 = 164096

typedef __attribute__((ext_vector_type(16))) __bf16 v16bf;
typedef __attribute__((ext_vector_type(8)))  float  v8f;

// ---------------- TDM support (guarded; falls back to global mask reads) ------
#if __has_builtin(__builtin_amdgcn_tensor_load_to_lds) && \
    __has_builtin(__builtin_amdgcn_s_wait_tensorcnt)
#define HAVE_TDM 1
typedef unsigned int v4u_t __attribute__((ext_vector_type(4)));
typedef int v8i_t __attribute__((ext_vector_type(8)));
typedef int v4i_t __attribute__((ext_vector_type(4)));

// DMA a 16-row x 2048-byte 2D tile (the attention-mask stripe) into LDS.
// D# layout per CDNA5 ISA ch.8 (group0 128b, group1 256b; groups 2/3 NULL-equiv).
__device__ __forceinline__ void tdm_load_mask_tile(const unsigned char* gsrc,
                                                   unsigned lds_byte_off) {
  unsigned long long ga = (unsigned long long)(uintptr_t)gsrc;
  v4u_t g0;
  g0[0] = 1u;                                   // count=1, user mode, no gather
  g0[1] = lds_byte_off;                         // lds_addr [63:32]
  g0[2] = (unsigned)(ga & 0xFFFFFFFFu);         // global_addr[31:0]  -> bits 95:64
  g0[3] = (unsigned)((ga >> 32) & 0x01FFFFFFu)  // global_addr[56:32] -> bits 120:96
        | (2u << 30);                           // type=2 ("image")   -> bits 127:126
  v8i_t g1;
  g1[0] = 0;                        // wg_mask=0, data_size=0 (1B), no pad/iterate
  g1[1] = (int)(2048u << 16);       // tensor_dim0[15:0]=2048 @ bits 63:48
  g1[2] = (int)(16u << 16);         // tensor_dim0 hi=0; tensor_dim1[15:0]=16 @ 95:80
  g1[3] = (int)(2048u << 16);       // tensor_dim1 hi=0; tile_dim0=2048 @ 127:112
  g1[4] = 16;                       // tile_dim1=16, tile_dim2=0
  g1[5] = 2048;                     // tensor_dim0_stride[31:0] (bytes)
  g1[6] = 0;                        // stride hi / dim1_stride lo
  g1[7] = 0;
  v4i_t z4 = {0, 0, 0, 0};
#if __clang_major__ >= 23
  v8i_t z8 = {0, 0, 0, 0, 0, 0, 0, 0};
  __builtin_amdgcn_tensor_load_to_lds(g0, g1, z4, z4, z8, 0);
#else
  __builtin_amdgcn_tensor_load_to_lds(g0, g1, z4, z4, 0);
#endif
}
#else
#define HAVE_TDM 0
#endif

// ---------------- WMMA fragment helpers (wave32, 16x16x32 bf16) ----------------
// A 16x32 (MxK), bf16 row-major source; caller points at (row0, kbase).
// lanes 0-15: M=lane, K 0-7 & 16-23 ; lanes 16-31: M=lane-16, K 8-15 & 24-31.
__device__ __forceinline__ v16bf load_a(const __bf16* A, int lda, int lane) {
  int m  = lane & 15;
  int k0 = (lane >> 4) << 3;           // 0 or 8
  const __bf16* p = A + m * lda + k0;
  union { v16bf v; uint4 u[2]; } r;
  r.u[0] = *(const uint4*)(p);         // K +0..7
  r.u[1] = *(const uint4*)(p + 16);    // K +16..23
  return r.v;
}

// Same A fragment but from an f32 source, converting to bf16 on the fly.
__device__ __forceinline__ v16bf load_a_f32(const float* A, int lda, int lane) {
  int m  = lane & 15;
  int k0 = (lane >> 4) << 3;
  const float* p = A + m * lda + k0;
  v16bf r;
#pragma unroll
  for (int t = 0; t < 8; ++t) r[t]     = (__bf16)p[t];
#pragma unroll
  for (int t = 0; t < 8; ++t) r[8 + t] = (__bf16)p[16 + t];
  return r;
}

// B 32x16 (KxN) where the source is stored transposed (row-major N x K, i.e.
// b[e] = Src[n][k]) -> two contiguous 16B loads per lane.
// lanes 0-15: N=lane, K 0-7 & 16-23 ; lanes 16-31: N=lane-16, K 8-15 & 24-31.
__device__ __forceinline__ v16bf load_bt(const __bf16* Bt, int ldt, int lane) {
  int n  = lane & 15;
  int kb = (lane >> 4) << 3;           // 0 or 8
  const __bf16* p = Bt + (size_t)n * ldt + kb;
  union { v16bf v; uint4 u[2]; } r;
  r.u[0] = *(const uint4*)(p);
  r.u[1] = *(const uint4*)(p + 16);
  return r.v;
}

__device__ __forceinline__ v8f wmma_bf16(v16bf a, v16bf b, v8f c) {
  return __builtin_amdgcn_wmma_f32_16x16x32_bf16(false, a, false, b, (short)0, c,
                                                 false, false);
}

// ---------------- staging kernels ----------------
__global__ void __launch_bounds__(256) cvt_f32_bf16(const float* __restrict__ s,
                                                    __bf16* __restrict__ d, int n) {
  int i = blockIdx.x * blockDim.x + threadIdx.x;
  if (i < n) d[i] = (__bf16)s[i];
}

// W [K,N] f32 row-major -> Wt [N,K] bf16 (so GEMM B-operands load contiguously)
__global__ void __launch_bounds__(256) cvt_transpose(const float* __restrict__ s,
                                                     __bf16* __restrict__ d) {
  int i = blockIdx.x * blockDim.x + threadIdx.x;   // over K*N
  int k = i >> 10;
  int n = i & 1023;
  d[(size_t)n * D_MODEL + k] = (__bf16)s[i];
}

// ---------------- GEMM kernels ----------------
// Out[M,N] = X[M,K] @ Wt^T. block=128 (4 waves); wave computes 16 rows x 64 cols
// with 4 independent accumulator chains sharing one A fragment.
__global__ void __launch_bounds__(128) proj_gemm(const __bf16* __restrict__ X,
                                                 const __bf16* __restrict__ Wt,
                                                 __bf16* __restrict__ Out) {
  int lane = threadIdx.x & 31, wave = threadIdx.x >> 5;
  int row0 = blockIdx.y * 16;
  int col0 = blockIdx.x * 256 + wave * 64;
  v8f a0 = {}, a1 = {}, a2 = {}, a3 = {};
#pragma unroll 2
  for (int k = 0; k < D_MODEL; k += 32) {
    v16bf a = load_a(X + (size_t)row0 * D_MODEL + k, D_MODEL, lane);
    if (k + 32 < D_MODEL)
      __builtin_prefetch(X + (size_t)row0 * D_MODEL + k + 32, 0, 1);
    v16bf b0 = load_bt(Wt + (size_t)(col0 +  0) * D_MODEL + k, D_MODEL, lane);
    v16bf b1 = load_bt(Wt + (size_t)(col0 + 16) * D_MODEL + k, D_MODEL, lane);
    v16bf b2 = load_bt(Wt + (size_t)(col0 + 32) * D_MODEL + k, D_MODEL, lane);
    v16bf b3 = load_bt(Wt + (size_t)(col0 + 48) * D_MODEL + k, D_MODEL, lane);
    a0 = wmma_bf16(a, b0, a0);
    a1 = wmma_bf16(a, b1, a1);
    a2 = wmma_bf16(a, b2, a2);
    a3 = wmma_bf16(a, b3, a3);
  }
  int n = col0 + (lane & 15);
  int m = row0 + ((lane >> 4) << 3);
  v8f accs[4] = {a0, a1, a2, a3};
#pragma unroll
  for (int t = 0; t < 4; ++t)
#pragma unroll
    for (int v = 0; v < 8; ++v)
      Out[(size_t)(m + v) * D_MODEL + (n + 16 * t)] = (__bf16)accs[t][v];
}

// Same GEMM but stores the result TRANSPOSED per batch:
// Vt[b][n][s] (n = 0..1023 projected dim, s = seq) so the context GEMM's
// B-operand reads contiguously. Each lane's 8 outputs are consecutive seq
// positions -> packed single 16B store.
__global__ void __launch_bounds__(128) proj_gemm_vt(const __bf16* __restrict__ X,
                                                    const __bf16* __restrict__ Wt,
                                                    __bf16* __restrict__ Vt) {
  int lane = threadIdx.x & 31, wave = threadIdx.x >> 5;
  int row0 = blockIdx.y * 16;
  int col0 = blockIdx.x * 256 + wave * 64;
  v8f a0 = {}, a1 = {}, a2 = {}, a3 = {};
#pragma unroll 2
  for (int k = 0; k < D_MODEL; k += 32) {
    v16bf a = load_a(X + (size_t)row0 * D_MODEL + k, D_MODEL, lane);
    v16bf b0 = load_bt(Wt + (size_t)(col0 +  0) * D_MODEL + k, D_MODEL, lane);
    v16bf b1 = load_bt(Wt + (size_t)(col0 + 16) * D_MODEL + k, D_MODEL, lane);
    v16bf b2 = load_bt(Wt + (size_t)(col0 + 32) * D_MODEL + k, D_MODEL, lane);
    v16bf b3 = load_bt(Wt + (size_t)(col0 + 48) * D_MODEL + k, D_MODEL, lane);
    a0 = wmma_bf16(a, b0, a0);
    a1 = wmma_bf16(a, b1, a1);
    a2 = wmma_bf16(a, b2, a2);
    a3 = wmma_bf16(a, b3, a3);
  }
  int m = row0 + ((lane >> 4) << 3);           // global row = b*SEQ + s
  int b = m >> 11;
  int s = m & (SEQ - 1);
  v8f accs[4] = {a0, a1, a2, a3};
#pragma unroll
  for (int t = 0; t < 4; ++t) {
    int n = col0 + 16 * t + (lane & 15);
    union { __bf16 h[8]; uint4 u; } pk;
#pragma unroll
    for (int v = 0; v < 8; ++v) pk.h[v] = (__bf16)accs[t][v];
    *(uint4*)(Vt + ((size_t)(b * D_MODEL + n)) * SEQ + s) = pk.u;
  }
}

// Fused scores + mask + softmax. block=256 (8 waves) handles 16 q-rows x 2048 keys.
// Scores staged in a 16 x STILE LDS tile (dynamic, 128KB) then row-softmaxed in
// place. The mask stripe is DMAed into LDS by the Tensor Data Mover concurrently
// with the WMMA phase (falls back to global reads if TDM builtin unavailable).
__global__ void __launch_bounds__(256) attn_scores_softmax(
    const __bf16* __restrict__ Qbf, const __bf16* __restrict__ Kbf,
    const unsigned char* __restrict__ mask, float* __restrict__ attn) {
  extern __shared__ float tile[];
  int lane = threadIdx.x & 31, wave = threadIdx.x >> 5;
  int bh = blockIdx.x;
  int b = bh >> 4, h = bh & 15;
  int q0 = blockIdx.y * 16;

#if HAVE_TDM
  if (wave == 0) {
    unsigned lds_off = (unsigned)(uintptr_t)tile + MASK_LDS_OFF; // flat low 32b = LDS offset
    tdm_load_mask_tile(mask + ((size_t)b * SEQ + q0) * SEQ, lds_off);
  }
#endif

  const __bf16* Qbase = Qbf + (size_t)(b * SEQ + q0) * D_MODEL + h * HEAD_DIM;
  const __bf16* Kbase = Kbf + (size_t)(b * SEQ) * D_MODEL + h * HEAD_DIM;

  v16bf qa0 = load_a(Qbase, D_MODEL, lane);        // K 0..31 of head
  v16bf qa1 = load_a(Qbase + 32, D_MODEL, lane);   // K 32..63

#pragma unroll 2
  for (int tt = 0; tt < 16; ++tt) {
    int col0 = (wave * 16 + tt) << 4;              // wave owns 16 contiguous tiles
    v16bf b0 = load_bt(Kbase + (size_t)col0 * D_MODEL, D_MODEL, lane);
    v16bf b1 = load_bt(Kbase + (size_t)col0 * D_MODEL + 32, D_MODEL, lane);
    v8f acc = {};
    acc = wmma_bf16(qa0, b0, acc);
    acc = wmma_bf16(qa1, b1, acc);
    int nl = col0 + (lane & 15);
    int mb = (lane >> 4) << 3;
#pragma unroll
    for (int v = 0; v < 8; ++v) tile[(mb + v) * STILE + nl] = acc[v];
  }
#if HAVE_TDM
  if (wave == 0) __builtin_amdgcn_s_wait_tensorcnt(0);  // mask DMA complete
#endif
  __syncthreads();

  // row softmax: 16 threads per row (row = tid>>4), xor-shuffles stay in half-wave
  int r = threadIdx.x >> 4;
  int g = threadIdx.x & 15;
#if HAVE_TDM
  const unsigned char* mrow =
      (const unsigned char*)tile + MASK_LDS_OFF + (size_t)r * SEQ;
#else
  const unsigned char* mrow = mask + ((size_t)b * SEQ + (q0 + r)) * SEQ;
#endif
  float* trow = tile + r * STILE;

  float mx = -3.0e38f;
  for (int j = 0; j < SEQ / 16; ++j) {
    int c = g + (j << 4);
    float s = mrow[c] ? -1.0e9f : trow[c] * 0.125f;   // /sqrt(64), mask==true -> -1e9
    trow[c] = s;
    mx = fmaxf(mx, s);
  }
#pragma unroll
  for (int o = 8; o >= 1; o >>= 1) mx = fmaxf(mx, __shfl_xor(mx, o, 32));

  float sum = 0.f;
  for (int j = 0; j < SEQ / 16; ++j) {
    int c = g + (j << 4);
    float e = __expf(trow[c] - mx);
    trow[c] = e;
    sum += e;
  }
#pragma unroll
  for (int o = 8; o >= 1; o >>= 1) sum += __shfl_xor(sum, o, 32);
  float inv = 1.0f / sum;

  float* arow = attn + ((size_t)bh * SEQ + (q0 + r)) * SEQ;
  for (int j = 0; j < SEQ / 16; ++j) {
    int c = g + (j << 4);
    arow[c] = trow[c] * inv;
  }
}

// context[bh] = attn[bh] (2048x2048 f32) @ V_bh. block=128 (4 waves); each wave
// owns a 16-row q-tile and all 64 head dims (4 accumulators). V read from the
// transposed Vt layout -> contiguous B loads.
__global__ void __launch_bounds__(128) attn_context(const float* __restrict__ attn,
                                                    const __bf16* __restrict__ Vt,
                                                    __bf16* __restrict__ Ctx) {
  int lane = threadIdx.x & 31, wave = threadIdx.x >> 5;
  int bh = blockIdx.x;
  int b = bh >> 4, h = bh & 15;
  int q0 = blockIdx.y * 64 + wave * 16;
  const float*  Abase = attn + ((size_t)bh * SEQ + q0) * SEQ;
  const __bf16* Vbase = Vt + ((size_t)(b * D_MODEL + h * HEAD_DIM)) * SEQ;
  v8f a0 = {}, a1 = {}, a2 = {}, a3 = {};
#pragma unroll 2
  for (int k = 0; k < SEQ; k += 32) {
    v16bf a = load_a_f32(Abase + k, SEQ, lane);
    if (k + 32 < SEQ) __builtin_prefetch(Abase + k + 32, 0, 1);
    v16bf b0 = load_bt(Vbase + (size_t)0  * SEQ + k, SEQ, lane);
    v16bf b1 = load_bt(Vbase + (size_t)16 * SEQ + k, SEQ, lane);
    v16bf b2 = load_bt(Vbase + (size_t)32 * SEQ + k, SEQ, lane);
    v16bf b3 = load_bt(Vbase + (size_t)48 * SEQ + k, SEQ, lane);
    a0 = wmma_bf16(a, b0, a0);
    a1 = wmma_bf16(a, b1, a1);
    a2 = wmma_bf16(a, b2, a2);
    a3 = wmma_bf16(a, b3, a3);
  }
  int m = b * SEQ + q0 + ((lane >> 4) << 3);
  v8f accs[4] = {a0, a1, a2, a3};
#pragma unroll
  for (int t = 0; t < 4; ++t) {
    int n = h * HEAD_DIM + 16 * t + (lane & 15);
#pragma unroll
    for (int v = 0; v < 8; ++v)
      Ctx[(size_t)(m + v) * D_MODEL + n] = (__bf16)accs[t][v];
  }
}

// Y = Ctx @ W_fc + residual (f32 out, pre-LayerNorm); W_fc pre-transposed.
__global__ void __launch_bounds__(128) out_proj(const __bf16* __restrict__ Ctx,
                                                const __bf16* __restrict__ Wt,
                                                const float* __restrict__ resid,
                                                float* __restrict__ Y) {
  int lane = threadIdx.x & 31, wave = threadIdx.x >> 5;
  int row0 = blockIdx.y * 16;
  int col0 = blockIdx.x * 256 + wave * 64;
  v8f a0 = {}, a1 = {}, a2 = {}, a3 = {};
#pragma unroll 2
  for (int k = 0; k < D_MODEL; k += 32) {
    v16bf a = load_a(Ctx + (size_t)row0 * D_MODEL + k, D_MODEL, lane);
    v16bf b0 = load_bt(Wt + (size_t)(col0 +  0) * D_MODEL + k, D_MODEL, lane);
    v16bf b1 = load_bt(Wt + (size_t)(col0 + 16) * D_MODEL + k, D_MODEL, lane);
    v16bf b2 = load_bt(Wt + (size_t)(col0 + 32) * D_MODEL + k, D_MODEL, lane);
    v16bf b3 = load_bt(Wt + (size_t)(col0 + 48) * D_MODEL + k, D_MODEL, lane);
    a0 = wmma_bf16(a, b0, a0);
    a1 = wmma_bf16(a, b1, a1);
    a2 = wmma_bf16(a, b2, a2);
    a3 = wmma_bf16(a, b3, a3);
  }
  int m = row0 + ((lane >> 4) << 3);
  v8f accs[4] = {a0, a1, a2, a3};
#pragma unroll
  for (int t = 0; t < 4; ++t) {
    int n = col0 + 16 * t + (lane & 15);
#pragma unroll
    for (int v = 0; v < 8; ++v) {
      size_t idx = (size_t)(m + v) * D_MODEL + n;
      Y[idx] = accs[t][v] + resid[idx];
    }
  }
}

__global__ void __launch_bounds__(256) layernorm_rows(const float* __restrict__ Y,
                                                      float* __restrict__ Out) {
  __shared__ float red[16];
  int row = blockIdx.x;
  const float* y = Y + (size_t)row * D_MODEL;
  float vals[4];
  float s = 0.f, ss = 0.f;
#pragma unroll
  for (int i = 0; i < 4; ++i) {
    float v = y[threadIdx.x + i * 256];
    vals[i] = v; s += v; ss += v * v;
  }
#pragma unroll
  for (int o = 16; o >= 1; o >>= 1) { s += __shfl_xor(s, o, 32); ss += __shfl_xor(ss, o, 32); }
  int wave = threadIdx.x >> 5, lane = threadIdx.x & 31;
  if (lane == 0) { red[wave] = s; red[8 + wave] = ss; }
  __syncthreads();
  if (threadIdx.x == 0) {
    float ts = 0.f, tss = 0.f;
    for (int w = 0; w < 8; ++w) { ts += red[w]; tss += red[8 + w]; }
    red[0] = ts; red[8] = tss;
  }
  __syncthreads();
  float mu  = red[0] / (float)D_MODEL;
  float var = red[8] / (float)D_MODEL - mu * mu;
  float inv = rsqrtf(var + LN_EPS);
  float* o = Out + (size_t)row * D_MODEL;
#pragma unroll
  for (int i = 0; i < 4; ++i) o[threadIdx.x + i * 256] = (vals[i] - mu) * inv;
}

// ---------------- launch ----------------
extern "C" void kernel_launch(void* const* d_in, const int* in_sizes, int n_in,
                              void* d_out, int out_size, void* d_ws, size_t ws_size,
                              hipStream_t stream) {
  const float* in_q = (const float*)d_in[0];
  const float* in_k = (const float*)d_in[1];
  const float* in_v = (const float*)d_in[2];
  const unsigned char* mask = (const unsigned char*)d_in[3];  // jax bool = 1 byte
  const float* Wq  = (const float*)d_in[4];
  const float* Wk  = (const float*)d_in[5];
  const float* Wv  = (const float*)d_in[6];
  const float* Wfc = (const float*)d_in[7];

  float* out  = (float*)d_out;                      // [4096,1024] LN output
  float* attn = out + (size_t)ROWS * D_MODEL;       // [2,16,2048,2048] attn probs

  char* ws = (char*)d_ws;                           // ~84 MB used
  const size_t ACT = (size_t)ROWS * D_MODEL * 2;    // 8 MiB bf16 activation
  const size_t WSZ = (size_t)D_MODEL * D_MODEL * 2; // 2 MiB bf16 weight
  __bf16* Xq  = (__bf16*)(ws);
  __bf16* Xk  = (__bf16*)(ws + ACT);
  __bf16* Xv  = (__bf16*)(ws + 2 * ACT);
  __bf16* Qb  = (__bf16*)(ws + 3 * ACT);
  __bf16* Kb  = (__bf16*)(ws + 4 * ACT);
  __bf16* Vt  = (__bf16*)(ws + 5 * ACT);            // [B,1024,2048] transposed V
  __bf16* Cx  = (__bf16*)(ws + 6 * ACT);
  __bf16* Wqt = (__bf16*)(ws + 7 * ACT);            // transposed bf16 weights [N,K]
  __bf16* Wkt = (__bf16*)(ws + 7 * ACT + WSZ);
  __bf16* Wvt = (__bf16*)(ws + 7 * ACT + 2 * WSZ);
  __bf16* Wft = (__bf16*)(ws + 7 * ACT + 3 * WSZ);
  float*  Yf  = (float*)(ws + 7 * ACT + 4 * WSZ);

  const int NA = ROWS * D_MODEL;       // 4,194,304
  const int NW = D_MODEL * D_MODEL;    // 1,048,576

  // 1) bf16 staging (activations straight, weights transposed)
  cvt_f32_bf16<<<NA / 256, 256, 0, stream>>>(in_q, Xq, NA);
  cvt_f32_bf16<<<NA / 256, 256, 0, stream>>>(in_k, Xk, NA);
  cvt_f32_bf16<<<NA / 256, 256, 0, stream>>>(in_v, Xv, NA);
  cvt_transpose<<<NW / 256, 256, 0, stream>>>(Wq,  Wqt);
  cvt_transpose<<<NW / 256, 256, 0, stream>>>(Wk,  Wkt);
  cvt_transpose<<<NW / 256, 256, 0, stream>>>(Wv,  Wvt);
  cvt_transpose<<<NW / 256, 256, 0, stream>>>(Wfc, Wft);

  // 2) Q/K/V projections (WMMA bf16); V stored transposed for the context GEMM
  dim3 gp(D_MODEL / 256, ROWS / 16);
  proj_gemm   <<<gp, 128, 0, stream>>>(Xq, Wqt, Qb);
  proj_gemm   <<<gp, 128, 0, stream>>>(Xk, Wkt, Kb);
  proj_gemm_vt<<<gp, 128, 0, stream>>>(Xv, Wvt, Vt);

  // 3) fused scores + mask + softmax -> attn (f32, straight to d_out);
  //    mask stripe DMAed into LDS by the TDM while WMMAs run
  dim3 gs(BATCH * N_HEADS, SEQ / 16);
  attn_scores_softmax<<<gs, 256, (size_t)SCORES_LDS_BYTES, stream>>>(
      Qb, Kb, mask, attn);

  // 4) context = attn @ V (WMMA bf16, attn converted on the fly)
  dim3 gc(BATCH * N_HEADS, SEQ / 64);
  attn_context<<<gc, 128, 0, stream>>>(attn, Vt, Cx);

  // 5) out-proj + residual, then 6) LayerNorm
  out_proj<<<gp, 128, 0, stream>>>(Cx, Wft, in_q, Yf);
  layernorm_rows<<<ROWS, 256, 0, stream>>>(Yf, out);
}
